// CrossDomainAttention_57690000720288
// MI455X (gfx1250) — matrix operand order repaired and checked by reading, not compiled, not run on hardware
//
#include <hip/hip_runtime.h>

typedef __attribute__((ext_vector_type(16))) __bf16 v16bf;
typedef __attribute__((ext_vector_type(8)))  float  v8f;
typedef __attribute__((ext_vector_type(4)))  int    v4i;
typedef __attribute__((ext_vector_type(4)))  float  v4f;

#define B_ 4
#define S_ 2048
#define E_ 2048
#define H_ 16
#define D_ 128

#define EP_F32  0
#define EP_BF16 1
#define EP_MIX  2

union Frag16 { v4i i4[2]; v16bf v; };

// ---------------------------------------------------------------------------
// Generic GEMM: C[M,N] = A[M,K] @ W[N,K]^T + bias[N]
// fp32 in memory -> bf16 LDS tiles (double buffered) -> f32 WMMA accumulate.
// Requires M%128==0, N%128==0, K%32==0 (true for all call sites).
// 256 threads = 8 waves; block tile 128x128; wave tile 64x32 (4x2 WMMA tiles).
// Software pipeline: global loads for tile k+1 issued before WMMA on tile k,
// converted/stored to the alternate LDS buffer after; one barrier per k-step.
// ---------------------------------------------------------------------------
__global__ __launch_bounds__(256) void gemm_bf16_kernel(
    const float* __restrict__ A, const float* __restrict__ W,
    const float* __restrict__ bias, void* __restrict__ Cout,
    int M, int N, int K, int ep,
    const float* __restrict__ mixBase, const float* __restrict__ gPtr, int rowsPerB)
{
    __shared__ __align__(16) __bf16 As[2][128][40];   // 32 K + 8 pad (80B rows)
    __shared__ __align__(16) __bf16 Ws[2][128][40];

    const int blockN = blockIdx.x * 128;
    const int blockM = blockIdx.y * 128;
    const int lane = threadIdx.x & 31;
    const int wid  = threadIdx.x >> 5;
    const int wm   = wid & 1;            // 2 waves over M -> 64 rows each
    const int wn   = wid >> 1;           // 4 waves over N -> 32 cols each
    const int m16  = lane & 15;
    const int hh   = lane >> 4;

    const int tcol = (threadIdx.x & 7) * 4;   // 0..28
    const int trow = threadIdx.x >> 3;        // 0..31

    const float* Aload = A + (size_t)(blockM + trow) * K + tcol;
    const float* Wload = W + (size_t)(blockN + trow) * K + tcol;

    v8f acc[4][2] = {};
    v4f ar[4], wr[4];

    // --- stage 0 ---
    #pragma unroll
    for (int j = 0; j < 4; ++j) {
        ar[j] = *(const v4f*)(Aload + (size_t)j * 32 * K);
        wr[j] = *(const v4f*)(Wload + (size_t)j * 32 * K);
    }
    #pragma unroll
    for (int j = 0; j < 4; ++j) {
        const int r = trow + j * 32;
        As[0][r][tcol + 0] = (__bf16)ar[j].x;
        As[0][r][tcol + 1] = (__bf16)ar[j].y;
        As[0][r][tcol + 2] = (__bf16)ar[j].z;
        As[0][r][tcol + 3] = (__bf16)ar[j].w;
        Ws[0][r][tcol + 0] = (__bf16)wr[j].x;
        Ws[0][r][tcol + 1] = (__bf16)wr[j].y;
        Ws[0][r][tcol + 2] = (__bf16)wr[j].z;
        Ws[0][r][tcol + 3] = (__bf16)wr[j].w;
    }

    int buf = 0;
    for (int k0 = 0; k0 < K; k0 += 32) {
        __syncthreads();                      // buf ready; other buf free
        const bool more = (k0 + 32) < K;      // uniform
        if (more) {                           // issue next tile's loads early
            #pragma unroll
            for (int j = 0; j < 4; ++j) {
                ar[j] = *(const v4f*)(Aload + (size_t)j * 32 * K + k0 + 32);
                wr[j] = *(const v4f*)(Wload + (size_t)j * 32 * K + k0 + 32);
            }
        }

        // A fragments: 16x32 bf16. lane m=L%16, h=L/16; elem i -> K = h*8+i(+8 hi)
        Frag16 af[4];
        #pragma unroll
        for (int mt = 0; mt < 4; ++mt) {
            const v4i* row = (const v4i*)&As[buf][wm * 64 + mt * 16 + m16][0];
            af[mt].i4[0] = row[hh];
            af[mt].i4[1] = row[hh + 2];
        }
        // B fragments: 32x16 bf16. lane n=L%16, h=L/16; elem i -> K = h*8+i(+8 hi)
        Frag16 bf_[2];
        #pragma unroll
        for (int nt = 0; nt < 2; ++nt) {
            const v4i* row = (const v4i*)&Ws[buf][wn * 32 + nt * 16 + m16][0];
            bf_[nt].i4[0] = row[hh];
            bf_[nt].i4[1] = row[hh + 2];
        }
        #pragma unroll
        for (int mt = 0; mt < 4; ++mt)
            #pragma unroll
            for (int nt = 0; nt < 2; ++nt)
                acc[mt][nt] = __builtin_amdgcn_wmma_f32_16x16x32_bf16(
                    false, af[mt].v, false, bf_[nt].v, (short)0, acc[mt][nt], false, false);

        if (more) {                           // convert + store into other buffer
            const int nb = buf ^ 1;
            #pragma unroll
            for (int j = 0; j < 4; ++j) {
                const int r = trow + j * 32;
                As[nb][r][tcol + 0] = (__bf16)ar[j].x;
                As[nb][r][tcol + 1] = (__bf16)ar[j].y;
                As[nb][r][tcol + 2] = (__bf16)ar[j].z;
                As[nb][r][tcol + 3] = (__bf16)ar[j].w;
                Ws[nb][r][tcol + 0] = (__bf16)wr[j].x;
                Ws[nb][r][tcol + 1] = (__bf16)wr[j].y;
                Ws[nb][r][tcol + 2] = (__bf16)wr[j].z;
                Ws[nb][r][tcol + 3] = (__bf16)wr[j].w;
            }
        }
        buf ^= 1;
    }

    float gv = 0.f;
    if (ep == EP_MIX) gv = gPtr[blockM / rowsPerB];   // uniform per block

    #pragma unroll
    for (int mt = 0; mt < 4; ++mt) {
        #pragma unroll
        for (int nt = 0; nt < 2; ++nt) {
            const int col     = blockN + wn * 32 + nt * 16 + m16;
            const int rowBase = blockM + wm * 64 + mt * 16 + hh * 8;
            const float bcol  = bias[col];
            #pragma unroll
            for (int r = 0; r < 8; ++r) {
                const size_t idx = (size_t)(rowBase + r) * N + col;
                const float v = acc[mt][nt][r] + bcol;
                if (ep == EP_F32) {
                    ((float*)Cout)[idx] = v;
                } else if (ep == EP_BF16) {
                    ((__bf16*)Cout)[idx] = (__bf16)v;
                } else { // EP_MIX: (1-g)*base + g*(adapter out)
                    const float bse = mixBase[idx];
                    ((__bf16*)Cout)[idx] = (__bf16)((1.0f - gv) * bse + gv * v);
                }
            }
        }
    }
}

// ---------------------------------------------------------------------------
// xmean[b,e] = mean over s of x[b,s,e]
// ---------------------------------------------------------------------------
__global__ __launch_bounds__(256) void colmean_kernel(const float* __restrict__ x,
                                                      float* __restrict__ xmean)
{
    const int idx = blockIdx.x * 256 + threadIdx.x;   // b*E + e
    const int b = idx / E_;
    const int e = idx % E_;
    const float* p = x + (size_t)b * S_ * E_ + e;
    float s0 = 0.f, s1 = 0.f, s2 = 0.f, s3 = 0.f;
    for (int t = 0; t < S_; t += 4) {
        s0 += p[(size_t)(t + 0) * E_];
        s1 += p[(size_t)(t + 1) * E_];
        s2 += p[(size_t)(t + 2) * E_];
        s3 += p[(size_t)(t + 3) * E_];
    }
    xmean[idx] = (s0 + s1 + s2 + s3) * (1.0f / (float)S_);
}

// ---------------------------------------------------------------------------
// g[b] = softmax(xmean[b] @ wg^T + bg)[1]
// ---------------------------------------------------------------------------
__global__ __launch_bounds__(256) void gate_kernel(const float* __restrict__ xmean,
                                                   const float* __restrict__ wg,
                                                   const float* __restrict__ bg,
                                                   float* __restrict__ g)
{
    __shared__ float s0[256], s1[256];
    const int b = blockIdx.x;
    float p0 = 0.f, p1 = 0.f;
    for (int e = threadIdx.x; e < E_; e += 256) {
        const float xm = xmean[b * E_ + e];
        p0 += xm * wg[e];
        p1 += xm * wg[E_ + e];
    }
    s0[threadIdx.x] = p0; s1[threadIdx.x] = p1;
    __syncthreads();
    for (int st = 128; st > 0; st >>= 1) {
        if (threadIdx.x < st) {
            s0[threadIdx.x] += s0[threadIdx.x + st];
            s1[threadIdx.x] += s1[threadIdx.x + st];
        }
        __syncthreads();
    }
    if (threadIdx.x == 0) {
        const float l0 = s0[0] + bg[0], l1 = s1[0] + bg[1];
        const float mx = fmaxf(l0, l1);
        const float e0 = __expf(l0 - mx), e1 = __expf(l1 - mx);
        g[b] = e1 / (e0 + e1);
    }
}

// ---------------------------------------------------------------------------
// In-place LayerNorm(128) + exact GELU, one wave (32 lanes x 4 elems) per row.
// ---------------------------------------------------------------------------
__global__ __launch_bounds__(256) void ln_gelu_kernel(float* __restrict__ t,
                                                      const float* __restrict__ gamma,
                                                      const float* __restrict__ beta)
{
    const int lane = threadIdx.x & 31;
    const int wid  = threadIdx.x >> 5;
    const size_t row = (size_t)blockIdx.x * 8 + wid;
    float* p = t + row * D_ + lane * 4;
    v4f x = *(const v4f*)p;
    float s = x.x + x.y + x.z + x.w;
    float q = x.x * x.x + x.y * x.y + x.z * x.z + x.w * x.w;
    #pragma unroll
    for (int m = 1; m <= 16; m <<= 1) { s += __shfl_xor(s, m); q += __shfl_xor(q, m); }
    const float mean = s * (1.0f / 128.0f);
    const float var  = q * (1.0f / 128.0f) - mean * mean;
    const float rstd = rsqrtf(var + 1e-5f);
    const float* gm = gamma + lane * 4;
    const float* bt = beta  + lane * 4;
    auto lg = [&](float xv, float g_, float b_) {
        const float y = (xv - mean) * rstd * g_ + b_;
        return 0.5f * y * (1.0f + erff(y * 0.70710678118654752f));
    };
    v4f y;
    y.x = lg(x.x, gm[0], bt[0]);
    y.y = lg(x.y, gm[1], bt[1]);
    y.z = lg(x.z, gm[2], bt[2]);
    y.w = lg(x.w, gm[3], bt[3]);
    *(v4f*)p = y;
}

// ---------------------------------------------------------------------------
// Flash attention. One block = (b,h, q-tile of 64). 4 waves, 16 q-rows/wave.
// qb/kb/vb: bf16 [B*S, E] (head h at column offset h*128). out: fp32 [B*S, E].
// ---------------------------------------------------------------------------
__global__ __launch_bounds__(128) void attn_kernel(const __bf16* __restrict__ qb,
                                                   const __bf16* __restrict__ kb,
                                                   const __bf16* __restrict__ vb,
                                                   float* __restrict__ outp)
{
    __shared__ __align__(16) __bf16 Ks[64][136];     // K tile [kv][d], 272B rows
    __shared__ __align__(16) __bf16 Vt[128][72];     // V tile transposed [d][kv]
    __shared__ __align__(16) __bf16 Ps[4][16][72];   // per-wave P staging

    const int lane = threadIdx.x & 31;
    const int wid  = threadIdx.x >> 5;
    const int m16  = lane & 15;
    const int hh   = lane >> 4;
    const int bh = blockIdx.y;
    const int b  = bh / H_;
    const int h  = bh % H_;
    const size_t rowBase = (size_t)b * S_;
    const size_t headOff = (size_t)h * D_;
    const int q0 = blockIdx.x * 64;

    // Q fragments for this wave's 16 rows, 4 k-chunks of 32 along d.
    Frag16 qf[4];
    {
        const int sq = q0 + wid * 16 + m16;
        const v4i* qrow = (const v4i*)(qb + (rowBase + sq) * E_ + headOff);
        #pragma unroll
        for (int kc = 0; kc < 4; ++kc) {
            qf[kc].i4[0] = qrow[kc * 4 + hh];
            qf[kc].i4[1] = qrow[kc * 4 + hh + 2];
        }
    }

    v8f Oacc[8] = {};
    float mstate[8], lstate[8];
    #pragma unroll
    for (int r = 0; r < 8; ++r) { mstate[r] = -1e30f; lstate[r] = 0.f; }
    const float scale = 0.08838834764831845f;   // 1/sqrt(128)

    for (int kv0 = 0; kv0 < S_; kv0 += 64) {
        __syncthreads();   // previous iteration's LDS reads done
        // K and V tiles: batched b128 global loads, then LDS stores.
        // 64 rows x 128 bf16 = 1024 b128 chunks per tile; 128 threads, 2 halves.
        #pragma unroll
        for (int half = 0; half < 2; ++half) {
            v4i kreg[4], vreg[4];
            #pragma unroll
            for (int i = 0; i < 4; ++i) {
                const int c = threadIdx.x + (half * 4 + i) * 128;
                const int row = c >> 4, ch = c & 15;
                const v4i* ksrc = (const v4i*)(kb + (rowBase + kv0 + row) * E_ + headOff);
                const v4i* vsrc = (const v4i*)(vb + (rowBase + kv0 + row) * E_ + headOff);
                kreg[i] = ksrc[ch];
                vreg[i] = vsrc[ch];
            }
            #pragma unroll
            for (int i = 0; i < 4; ++i) {
                const int c = threadIdx.x + (half * 4 + i) * 128;
                const int row = c >> 4, ch = c & 15;
                ((v4i*)&Ks[row][0])[ch] = kreg[i];
                union { v4i q; __bf16 e[8]; } u;
                u.q = vreg[i];
                #pragma unroll
                for (int j = 0; j < 8; ++j) Vt[ch * 8 + j][row] = u.e[j];   // transpose
            }
        }
        __syncthreads();

        // S = Q @ K^T : 4 n-tiles of 16 kv cols, k-dim = d = 4 chunks of 32
        v8f sc[4];
        #pragma unroll
        for (int nt = 0; nt < 4; ++nt) {
            v8f a = {};
            const v4i* krow = (const v4i*)&Ks[nt * 16 + m16][0];
            #pragma unroll
            for (int kc = 0; kc < 4; ++kc) {
                Frag16 kf;
                kf.i4[0] = krow[kc * 4 + hh];
                kf.i4[1] = krow[kc * 4 + hh + 2];
                a = __builtin_amdgcn_wmma_f32_16x16x32_bf16(
                        false, qf[kc].v, false, kf.v, (short)0, a, false, false);
            }
            #pragma unroll
            for (int r = 0; r < 8; ++r) sc[nt][r] = a[r] * scale;
        }

        // Online softmax (row = r + 8*hh; reduce over 16 lanes of the half-wave)
        float mnew[8], alpha[8], rsum[8];
        #pragma unroll
        for (int r = 0; r < 8; ++r) {
            float tm = fmaxf(fmaxf(sc[0][r], sc[1][r]), fmaxf(sc[2][r], sc[3][r]));
            #pragma unroll
            for (int m = 1; m <= 8; m <<= 1) tm = fmaxf(tm, __shfl_xor(tm, m));
            mnew[r]  = fmaxf(mstate[r], tm);
            alpha[r] = __expf(mstate[r] - mnew[r]);
            rsum[r]  = 0.f;
        }
        #pragma unroll
        for (int nt = 0; nt < 4; ++nt)
            #pragma unroll
            for (int r = 0; r < 8; ++r) {
                const float p = __expf(sc[nt][r] - mnew[r]);
                sc[nt][r] = p;
                rsum[r] += p;
            }
        #pragma unroll
        for (int r = 0; r < 8; ++r) {
            float t = rsum[r];
            #pragma unroll
            for (int m = 1; m <= 8; m <<= 1) t += __shfl_xor(t, m);
            lstate[r] = lstate[r] * alpha[r] + t;
            mstate[r] = mnew[r];
        }
        #pragma unroll
        for (int ot = 0; ot < 8; ++ot)
            #pragma unroll
            for (int r = 0; r < 8; ++r) Oacc[ot][r] *= alpha[r];

        // P (C layout f32) -> LDS -> A layout bf16 (same-wave LDS ops in order)
        #pragma unroll
        for (int nt = 0; nt < 4; ++nt)
            #pragma unroll
            for (int r = 0; r < 8; ++r)
                Ps[wid][r + 8 * hh][nt * 16 + m16] = (__bf16)sc[nt][r];

        Frag16 pf[2];
        const v4i* prow = (const v4i*)&Ps[wid][m16][0];
        #pragma unroll
        for (int kc = 0; kc < 2; ++kc) {
            pf[kc].i4[0] = prow[kc * 4 + hh];
            pf[kc].i4[1] = prow[kc * 4 + hh + 2];
        }
        // O += P @ V : 8 n-tiles along d, k-dim = kv = 2 chunks of 32
        #pragma unroll
        for (int ot = 0; ot < 8; ++ot) {
            const v4i* vrow = (const v4i*)&Vt[ot * 16 + m16][0];
            #pragma unroll
            for (int kc = 0; kc < 2; ++kc) {
                Frag16 vf;
                vf.i4[0] = vrow[kc * 4 + hh];
                vf.i4[1] = vrow[kc * 4 + hh + 2];
                Oacc[ot] = __builtin_amdgcn_wmma_f32_16x16x32_bf16(
                               false, pf[kc].v, false, vf.v, (short)0, Oacc[ot], false, false);
            }
        }
    }

    #pragma unroll
    for (int ot = 0; ot < 8; ++ot) {
        const int e = (int)headOff + ot * 16 + m16;
        #pragma unroll
        for (int r = 0; r < 8; ++r) {
            const int sq = q0 + wid * 16 + hh * 8 + r;
            outp[(rowBase + sq) * E_ + e] = Oacc[ot][r] / lstate[r];
        }
    }
}

// ---------------------------------------------------------------------------
extern "C" void kernel_launch(void* const* d_in, const int* in_sizes, int n_in,
                              void* d_out, int out_size, void* d_ws, size_t ws_size,
                              hipStream_t stream)
{
    const float* x   = (const float*)d_in[0];
    const float* ctx = (const float*)d_in[1];
    const float* wq  = (const float*)d_in[2];
    const float* bq  = (const float*)d_in[3];
    const float* wk  = (const float*)d_in[4];
    const float* bk  = (const float*)d_in[5];
    const float* wv  = (const float*)d_in[6];
    const float* bv  = (const float*)d_in[7];
    const float* wo  = (const float*)d_in[8];
    const float* bo  = (const float*)d_in[9];
    const float* wa1 = (const float*)d_in[10];
    const float* ba1 = (const float*)d_in[11];
    const float* lng = (const float*)d_in[12];
    const float* lnb = (const float*)d_in[13];
    const float* wa2 = (const float*)d_in[14];
    const float* ba2 = (const float*)d_in[15];
    const float* wg  = (const float*)d_in[16];
    const float* bg  = (const float*)d_in[17];
    float* out = (float*)d_out;

    char* ws = (char*)d_ws;
    size_t off = 0;
    auto carve = [&](size_t bytes) -> void* {
        void* p = ws + off;
        off += (bytes + 255) & ~(size_t)255;
        return p;
    };
    __bf16* qb  = (__bf16*)carve((size_t)B_ * S_ * E_ * 2);   // Q bf16
    __bf16* kbv = (__bf16*)carve((size_t)B_ * S_ * E_ * 2);   // mixed K bf16
    __bf16* vbv = (__bf16*)carve((size_t)B_ * S_ * E_ * 2);   // mixed V bf16
    float*  PF  = (float*)carve((size_t)B_ * S_ * E_ * 4);    // K/V projection (fp32)
    float*  T1  = (float*)carve((size_t)B_ * S_ * E_ * 4);    // adapter hidden / attn out
    float*  xmn = (float*)carve((size_t)B_ * E_ * 4);
    float*  gbf = (float*)carve(64);

    const dim3 blk(256);
    const int  MBS = B_ * S_;          // 8192
    const int  MAD = B_ * S_ * H_;     // 131072 adapter rows
    const dim3 gridP(E_ / 128, MBS / 128);   // projection GEMMs
    const dim3 gridA(D_ / 128, MAD / 128);   // adapter GEMMs (N=128)

    // gating
    colmean_kernel<<<dim3((B_ * E_) / 256), blk, 0, stream>>>(x, xmn);
    gate_kernel<<<dim3(B_), blk, 0, stream>>>(xmn, wg, bg, gbf);

    // Q projection -> bf16
    gemm_bf16_kernel<<<gridP, blk, 0, stream>>>(x, wq, bq, qb, MBS, E_, E_, EP_BF16,
                                                nullptr, gbf, 1);
    // K path: proj -> adapter1 -> LN+GELU -> adapter2 fused with gate mix
    gemm_bf16_kernel<<<gridP, blk, 0, stream>>>(ctx, wk, bk, PF, MBS, E_, E_, EP_F32,
                                                nullptr, gbf, 1);
    gemm_bf16_kernel<<<gridA, blk, 0, stream>>>(PF, wa1, ba1, T1, MAD, D_, D_, EP_F32,
                                                nullptr, gbf, 1);
    ln_gelu_kernel<<<dim3(MAD / 8), blk, 0, stream>>>(T1, lng, lnb);
    gemm_bf16_kernel<<<gridA, blk, 0, stream>>>(T1, wa2, ba2, kbv, MAD, D_, D_, EP_MIX,
                                                PF, gbf, S_ * H_);
    // V path
    gemm_bf16_kernel<<<gridP, blk, 0, stream>>>(ctx, wv, bv, PF, MBS, E_, E_, EP_F32,
                                                nullptr, gbf, 1);
    gemm_bf16_kernel<<<gridA, blk, 0, stream>>>(PF, wa1, ba1, T1, MAD, D_, D_, EP_F32,
                                                nullptr, gbf, 1);
    ln_gelu_kernel<<<dim3(MAD / 8), blk, 0, stream>>>(T1, lng, lnb);
    gemm_bf16_kernel<<<gridA, blk, 0, stream>>>(T1, wa2, ba2, vbv, MAD, D_, D_, EP_MIX,
                                                PF, gbf, S_ * H_);

    // flash attention -> T1 (fp32 [B*S, E], heads interleaved back into E)
    attn_kernel<<<dim3(S_ / 64, B_ * H_), dim3(128), 0, stream>>>(qb, kbv, vbv, T1);

    // output projection
    gemm_bf16_kernel<<<gridP, blk, 0, stream>>>(T1, wo, bo, out, MBS, E_, E_, EP_F32,
                                                nullptr, gbf, 1);
}
